// TPTGCN_33818572489415
// MI455X (gfx1250) — matrix operand order repaired and checked by reading, not compiled, not run on hardware
//
#include <hip/hip_runtime.h>

#define N_E    100000
#define N_R    20000
#define N_ALL  120000
#define DIMS   128
#define E_PRIM 1600000
#define E_RELA 1920000

#define TPB 256          // 8 waves
#define TPBLK 4          // row-tiles (of 16) per block -> 64 rows/block
#define FS  24           // fragment slot stride in ushorts (48B: b128-aligned, bank-spread)

typedef __attribute__((ext_vector_type(16))) __bf16 v16bf;
typedef __attribute__((ext_vector_type(8)))  float  v8f;

union Frag { uint4 u4[2]; v16bf v; };

// ---------- helpers ----------
__device__ __forceinline__ unsigned short f2bf_bits(float f) {
    union { float f; unsigned u; } x; x.f = f;
    unsigned r = x.u + 0x7FFFu + ((x.u >> 16) & 1u);   // round-to-nearest-even
    return (unsigned short)(r >> 16);
}

// Inverse of the documented 16-bit A 16x32 fragment layout:
// forward: half h (reg v=h/2, pair p=h&1) of lane-group g holds
//          k = (v<4?0:16) + g*8 + (v&3)*2 + p
// given k-within-32 (kl), return half index h and group g.
__device__ __forceinline__ int a_inv(int kl, int& g) {
    g = (kl >> 3) & 1;
    const int v = ((kl & 16) >> 2) + ((kl & 7) >> 1);
    return (v << 1) | (kl & 1);
}

// ---------------------------------------------------------------------------
// Unified tiled GEMM: C16x16 tiles of  A[M x 128] @ W[128 x 128]  via
// v_wmma_f32_16x16x32_bf16.  Both operands pre-swizzled into LDS in per-lane
// fragment order so the inner loop is pure ds_load_b128 -> v_wmma.
// HIGHWAY=false: C = A@W          (support GEMM)
// HIGHWAY=true : gate = sigmoid(A@W + bg); Out = gate*relu(Agg+bias) + (1-gate)*A
// ---------------------------------------------------------------------------
template <bool HIGHWAY>
__global__ __launch_bounds__(TPB)
void gemm128_kernel(const float* __restrict__ A, const float* __restrict__ W,
                    float* __restrict__ Out, const float* __restrict__ Agg,
                    const float* __restrict__ bg, const float* __restrict__ bias,
                    int M) {
    // A fragments: [tile t][kk][lane] slots   -> 4*4*32 slots
    // W fragments: [wave w][kk][lane] slots   -> 8*4*32 slots
    __shared__ unsigned short As_frag[TPBLK * 4 * 32 * FS];   // 24576 B
    __shared__ unsigned short Ws_frag[8 * 4 * 32 * FS];       // 49152 B

    const int tid   = threadIdx.x;
    const int m0blk = blockIdx.x * (TPBLK * 16);

    // ---- stage A (64 rows x 128), swizzled to fragment order ----
    for (int i = tid; i < TPBLK * 16 * 128; i += TPB) {
        const int mg = i >> 7;                // row within block (0..63)
        const int k  = i & 127;
        const int row = m0blk + mg;
        const unsigned short bits =
            (row < M) ? f2bf_bits(A[(size_t)row * DIMS + k]) : (unsigned short)0;
        const int t = mg >> 4, m = mg & 15;
        const int kk = k >> 5, kl = k & 31;
        const int g = kl >> 4; const int h = kl & 15;
        As_frag[(((t * 4 + kk) * 32) + g * 16 + m) * FS + h] = bits;
    }
    // ---- stage W (128 x 128), swizzled to B-fragment order ----
    // B order used here (32x16 bf16, same contiguous k order as A): lane = g*16 + n, half h -> K = g*16 + h
    for (int i = tid; i < 128 * 128; i += TPB) {
        const int k = i >> 7, col = i & 127;
        const unsigned short bits = f2bf_bits(W[i]);
        const int w = col >> 4, n = col & 15;
        const int kk = k >> 5, r = k & 31;
        const int g = r >> 4, h = r & 15;
        Ws_frag[(((w * 4 + kk) * 32) + g * 16 + n) * FS + h] = bits;
    }
    __syncthreads();

    const int wave = tid >> 5, lane = tid & 31;
    const int n0 = wave * 16;
    const int mL = lane & 15, gg = lane >> 4;

    float bgn = 0.0f, bn = 0.0f;
    if (HIGHWAY) { bgn = bg[n0 + mL]; bn = bias[n0 + mL]; }

    for (int t = 0; t < TPBLK; ++t) {
        const int m0 = m0blk + t * 16;
        if (m0 >= M) break;                    // uniform: EXEC all-1s at WMMAs

        v8f acc = {};
#pragma unroll
        for (int kk = 0; kk < 4; ++kk) {       // K = 4 * 32
            Frag fa, fb;
            const uint4* pa =
                (const uint4*)&As_frag[(((t * 4 + kk) * 32) + lane) * FS];
            const uint4* pb =
                (const uint4*)&Ws_frag[(((wave * 4 + kk) * 32) + lane) * FS];
            fa.u4[0] = pa[0]; fa.u4[1] = pa[1];    // 2x ds_load_b128
            fb.u4[0] = pb[0]; fb.u4[1] = pb[1];    // 2x ds_load_b128
            acc = __builtin_amdgcn_wmma_f32_16x16x32_bf16(
                      false, fa.v, false, fb.v, (short)0, acc, false, false);
        }

        // C/D layout: lane n = lane&15, VGPR r -> M = (lane>>4)*8 + r
#pragma unroll
        for (int r = 0; r < 8; ++r) {
            const int m = gg * 8 + r;
            const size_t idx = (size_t)(m0 + m) * DIMS + n0 + mL;
            if (HIGHWAY) {
                const float l1 = A[idx];
                float l2 = Agg[idx] + bn;
                l2 = l2 > 0.0f ? l2 : 0.0f;
                const float gate = 1.0f / (1.0f + __expf(-(acc[r] + bgn)));
                Out[idx] = gate * l2 + (1.0f - gate) * l1;
            } else {
                Out[idx] = acc[r];
            }
        }
    }
}

// ---------- SpMM: one wave32 per edge, hw f32 atomics ----------
__global__ __launch_bounds__(TPB)
void spmm_kernel(const int* __restrict__ row, const int* __restrict__ col,
                 const float* __restrict__ val, const float* __restrict__ X,
                 float* __restrict__ Y, int nE) {
    const int e    = (int)((blockIdx.x * (size_t)blockDim.x + threadIdx.x) >> 5);
    const int lane = threadIdx.x & 31;
    if (e >= nE) return;
    const int   r = row[e];
    const int   c = col[e];
    const float v = val[e];
    const float4 d = ((const float4*)(X + (size_t)c * DIMS))[lane];
    float* yp = Y + (size_t)r * DIMS + lane * 4;
    unsafeAtomicAdd(yp + 0, v * d.x);
    unsafeAtomicAdd(yp + 1, v * d.y);
    unsafeAtomicAdd(yp + 2, v * d.z);
    unsafeAtomicAdd(yp + 3, v * d.w);
}

extern "C" void kernel_launch(void* const* d_in, const int* in_sizes, int n_in,
                              void* d_out, int out_size, void* d_ws, size_t ws_size,
                              hipStream_t stream) {
    const float* e_x  = (const float*)d_in[0];
    const float* r_x  = (const float*)d_in[1];
    const float* W1   = (const float*)d_in[2];
    const float* b1   = (const float*)d_in[3];
    const float* Wm   = (const float*)d_in[4];   // weight_matrix (highway gate)
    const float* bg   = (const float*)d_in[5];   // bias_gate
    const float* W2   = (const float*)d_in[6];
    const float* b2   = (const float*)d_in[7];
    const int*   prow = (const int*)d_in[8];
    const int*   pcol = (const int*)d_in[9];
    const float* pval = (const float*)d_in[10];
    const int*   rrow = (const int*)d_in[11];
    const int*   rcol = (const int*)d_in[12];
    const float* rval = (const float*)d_in[13];

    float* cat     = (float*)d_ws;                    // [N_ALL x 128]
    float* support = cat + (size_t)N_ALL * DIMS;      // [N_ALL x 128]
    float* agg     = (float*)d_out;                   // SpMM accumulator (reused)

    const int blkE   = (N_E   + TPBLK * 16 - 1) / (TPBLK * 16);
    const int blkALL = (N_ALL + TPBLK * 16 - 1) / (TPBLK * 16);

    // ---- Layer 1 (entities only) ----
    gemm128_kernel<false><<<blkE, TPB, 0, stream>>>(e_x, W1, support,
                                                    nullptr, nullptr, nullptr, N_E);
    hipMemsetAsync(agg, 0, (size_t)N_E * DIMS * sizeof(float), stream);
    spmm_kernel<<<E_PRIM / 8, TPB, 0, stream>>>(prow, pcol, pval, support, agg, E_PRIM);
    gemm128_kernel<true><<<blkE, TPB, 0, stream>>>(e_x, Wm, cat,
                                                   agg, bg, b1, N_E);
    // cat tail = relation embeddings
    hipMemcpyAsync(cat + (size_t)N_E * DIMS, r_x,
                   (size_t)N_R * DIMS * sizeof(float),
                   hipMemcpyDeviceToDevice, stream);

    // ---- Layer 2 (entities + relations) ----
    gemm128_kernel<false><<<blkALL, TPB, 0, stream>>>(cat, W2, support,
                                                      nullptr, nullptr, nullptr, N_ALL);
    hipMemsetAsync(agg, 0, (size_t)N_ALL * DIMS * sizeof(float), stream);
    spmm_kernel<<<E_RELA / 8, TPB, 0, stream>>>(rrow, rcol, rval, support, agg, E_RELA);
    gemm128_kernel<true><<<blkALL, TPB, 0, stream>>>(cat, Wm, (float*)d_out,
                                                     agg, bg, b2, N_ALL);
}
